// TransformerEncoderLayer_32959579030272
// MI455X (gfx1250) — matrix operand order repaired and checked by reading, not compiled, not run on hardware
//
#include <hip/hip_runtime.h>
#include <math.h>

// ---------------------------------------------------------------------------
// Transformer-XL encoder layer for gfx1250 (MI455X), bf16 WMMA everywhere.
// S=1024 B=4 D=1024 NH=16 DH=64 DF=4096
// ---------------------------------------------------------------------------

typedef __bf16 bf16;
typedef __attribute__((ext_vector_type(16))) __bf16 v16bf;
typedef __attribute__((ext_vector_type(8)))  float  v8f;
typedef __attribute__((ext_vector_type(4)))  int    v4i_a;

#define S_LEN   1024
#define B_SZ    4
#define D_MODEL 1024
#define N_HEAD  16
#define D_HEAD  64
#define D_FF    4096
#define M_ROWS  4096   /* S*B   */
#define MP_ROWS 4100   /* (S+1)*B */
#define C_PITCH 1040   /* padded 1025 */

#if __has_builtin(__builtin_amdgcn_global_load_async_to_lds_b128)
#define HAVE_ASYNC_LDS 1
#define GLOBAL_AS __attribute__((address_space(1)))
#define LDS_AS    __attribute__((address_space(3)))
#endif

__device__ __forceinline__ void wait_async_then_none() {
#ifdef HAVE_ASYNC_LDS
#if __has_builtin(__builtin_amdgcn_s_wait_asynccnt)
  __builtin_amdgcn_s_wait_asynccnt(0);
#else
  asm volatile("s_wait_asynccnt 0" ::: "memory");
#endif
#endif
}

__device__ __forceinline__ v8f wmma_bf(v16bf a, v16bf b, v8f c) {
  return __builtin_amdgcn_wmma_f32_16x16x32_bf16(false, a, false, b, (short)0, c,
                                                 false, false);
}

// A-fragment (16x32, 16-bit): lane<16 -> K {0..15}, lane>=16 -> {16..31}
__device__ __forceinline__ v16bf load_a_frag(const bf16* rowp, int off) {
  v16bf r;
#pragma unroll
  for (int i = 0; i < 8; ++i) r[i] = rowp[off + i];
#pragma unroll
  for (int i = 0; i < 8; ++i) r[8 + i] = rowp[off + 8 + i];
  return r;
}

// B-fragment (32x16, 16-bit): lane reads 16 contiguous K values at column lane%16
__device__ __forceinline__ v16bf load_b_frag(const bf16* p) {
  v16bf r;
#pragma unroll
  for (int i = 0; i < 16; ++i) r[i] = p[i];
  return r;
}

// ---------------------------------------------------------------------------
// fp32 -> bf16 conversion (plain and transposed)
// ---------------------------------------------------------------------------
__global__ void cvt_kernel(const float* __restrict__ in, bf16* __restrict__ out,
                           int n) {
  int base = (blockIdx.x * 256 + threadIdx.x) * 4;
#pragma unroll
  for (int e = 0; e < 4; ++e) {
    int i = base + e;
    if (i < n) out[i] = (bf16)in[i];
  }
}

__global__ void cvtT_kernel(const float* __restrict__ in, bf16* __restrict__ out,
                            int R, int C) {
  int idx = blockIdx.x * 256 + threadIdx.x;
  if (idx < R * C) {
    int r = idx / C, c = idx - r * C;
    out[(size_t)c * R + r] = (bf16)in[idx];
  }
}

// ---------------------------------------------------------------------------
// Generic bf16 GEMM: C[M,N] = A[M,K] * B[K,N]  (fp32 accum)
// MODE 0: fp32 out (+bias)    1: bf16 out (+bias)   2: bf16 gelu(acc+bias)
// MODE 3: dual bf16 out (acc+bias1, acc+bias2)      4: bf16 V-transposed out
// A tile is staged with GLOBAL_LOAD_ASYNC_TO_LDS_B128 when available.
// ---------------------------------------------------------------------------
template <int MODE>
__global__ __launch_bounds__(256) void gemm_bf16_kernel(
    const bf16* __restrict__ A, const bf16* __restrict__ B, int M, int N, int K,
    const float* __restrict__ bias1, const float* __restrict__ bias2,
    void* __restrict__ out1, void* __restrict__ out2) {
  __shared__ __align__(16) bf16 As[128][48];
  __shared__ __align__(16) bf16 Bs[128][48];  // transposed: [n][k]

  const int tid = threadIdx.x;
  const int lane = tid & 31;
  const int w = tid >> 5;
  const int wm = w & 3;   // 4 waves along M
  const int wn = w >> 2;  // 2 waves along N
  const int m0 = blockIdx.y * 128;
  const int n0 = blockIdx.x * 128;
  const int lan15 = lane & 15;
  const int offA = (lane < 16) ? 0 : 16;
  const int offB = (lane < 16) ? 0 : 16;
  const int roff = (lane < 16) ? 0 : 8;

  v8f acc[2][4];
#pragma unroll
  for (int i = 0; i < 2; ++i)
#pragma unroll
    for (int j = 0; j < 4; ++j)
#pragma unroll
      for (int r = 0; r < 8; ++r) acc[i][j][r] = 0.0f;

  for (int k0 = 0; k0 < K; k0 += 32) {
    // stage A tile 128x32 (16B per thread x2); async DMA to LDS when available
#pragma unroll
    for (int r2 = 0; r2 < 2; ++r2) {
      int idx = tid + r2 * 256;
      int row = idx >> 2, seg = idx & 3;
      int gr = m0 + row;
      if (gr >= M) gr = M - 1;
      const bf16* sp = A + (size_t)gr * K + k0 + seg * 8;
      __builtin_prefetch(sp + 32, 0, 1);  // global_prefetch next K tile
#ifdef HAVE_ASYNC_LDS
      __builtin_amdgcn_global_load_async_to_lds_b128(
          (GLOBAL_AS v4i_a*)(void*)sp,
          (LDS_AS v4i_a*)(void*)&As[row][seg * 8], 0, 0);
#else
      *(uint4*)&As[row][seg * 8] = *(const uint4*)sp;
#endif
    }
    // stage B tile 32x128, stored transposed [n][k] (synchronous: transposes)
#pragma unroll
    for (int r2 = 0; r2 < 2; ++r2) {
      int idx = tid + r2 * 256;
      int kr = idx >> 4, seg = idx & 15;
      const bf16* sp = B + (size_t)(k0 + kr) * N + n0 + seg * 8;
      __builtin_prefetch(sp + (size_t)32 * N, 0, 1);
      __align__(16) bf16 tmp[8];
      *(uint4*)tmp = *(const uint4*)sp;
#pragma unroll
      for (int e = 0; e < 8; ++e) Bs[seg * 8 + e][kr] = tmp[e];
    }
    wait_async_then_none();  // this wave's async copies complete
    __syncthreads();

    v16bf af[2], bfr[4];
#pragma unroll
    for (int mt = 0; mt < 2; ++mt)
      af[mt] = load_a_frag(&As[wm * 32 + mt * 16 + lan15][0], offA);
#pragma unroll
    for (int nt = 0; nt < 4; ++nt)
      bfr[nt] = load_b_frag(&Bs[wn * 64 + nt * 16 + lan15][offB]);
#pragma unroll
    for (int mt = 0; mt < 2; ++mt)
#pragma unroll
      for (int nt = 0; nt < 4; ++nt)
        acc[mt][nt] = wmma_bf(af[mt], bfr[nt], acc[mt][nt]);
    __syncthreads();
  }

  // epilogue
#pragma unroll
  for (int mt = 0; mt < 2; ++mt) {
#pragma unroll
    for (int nt = 0; nt < 4; ++nt) {
      int col = n0 + wn * 64 + nt * 16 + lan15;
#pragma unroll
      for (int r = 0; r < 8; ++r) {
        int row = m0 + wm * 32 + mt * 16 + r + roff;
        if (row < M) {
          float v = acc[mt][nt][r];
          if (MODE == 0) {
            float bv = bias1 ? bias1[col] : 0.0f;
            ((float*)out1)[(size_t)row * N + col] = v + bv;
          } else if (MODE == 1) {
            float bv = bias1 ? bias1[col] : 0.0f;
            ((bf16*)out1)[(size_t)row * N + col] = (bf16)(v + bv);
          } else if (MODE == 2) {
            float t = v + bias1[col];
            float g = 0.5f * t * (1.0f + erff(t * 0.70710678118f));
            ((bf16*)out1)[(size_t)row * N + col] = (bf16)g;
          } else if (MODE == 3) {
            ((bf16*)out1)[(size_t)row * N + col] = (bf16)(v + bias1[col]);
            ((bf16*)out2)[(size_t)row * N + col] = (bf16)(v + bias2[col]);
          } else {  // MODE 4: store V transposed: vT[(b*NH+n)*DH+dh][i]
            ((bf16*)out1)[((size_t)(row & 3) * 1024 + col) * 1024 + (row >> 2)] =
                (bf16)v;
          }
        }
      }
    }
  }
}

// ---------------------------------------------------------------------------
// bd raw matrix: C[bn][i][p] = (q+r_r_bias)_i . k_pos_p   (K = 64, 2 WMMA steps)
// grid (64 bn, 16 i-blocks of 64, 17 p-blocks of 64), block 128 (4 waves)
// ---------------------------------------------------------------------------
__global__ __launch_bounds__(128) void bd_kernel(const bf16* __restrict__ qbB,
                                                 const bf16* __restrict__ kposb,
                                                 bf16* __restrict__ Cbd) {
  const int bn = blockIdx.x;
  const int b = bn >> 4, n = bn & 15;
  const int lane = threadIdx.x & 31;
  const int w = threadIdx.x >> 5;
  const int i0 = blockIdx.y * 64 + w * 16;
  const int p0base = blockIdx.z * 64;
  const int lan15 = lane & 15;
  const int offA = (lane < 16) ? 0 : 16;
  const int offB = (lane < 16) ? 0 : 16;
  const int roff = (lane < 16) ? 0 : 8;

  const bf16* qrow = qbB + ((size_t)(i0 + lan15) * B_SZ + b) * D_MODEL + n * D_HEAD;
  v16bf aq0 = load_a_frag(qrow, offA);
  v16bf aq1 = load_a_frag(qrow + 32, offA);

  bf16* cbase = Cbd + (size_t)bn * S_LEN * C_PITCH;
#pragma unroll
  for (int pt = 0; pt < 4; ++pt) {
    int p = p0base + pt * 16 + lan15;
    int pc = (p > 1024) ? 1024 : p;
    const bf16* krow = kposb + ((size_t)pc * B_SZ + b) * D_MODEL + n * D_HEAD;
    v8f acc;
#pragma unroll
    for (int r = 0; r < 8; ++r) acc[r] = 0.0f;
    acc = wmma_bf(aq0, load_b_frag(krow + offB), acc);
    acc = wmma_bf(aq1, load_b_frag(krow + 32 + offB), acc);
    if (p <= 1024) {
#pragma unroll
      for (int r = 0; r < 8; ++r) {
        int i = i0 + r + roff;
        cbase[(size_t)i * C_PITCH + p] = (bf16)acc[r];
      }
    }
  }
}

// ---------------------------------------------------------------------------
// Flash attention with rel_shift read of Cbd.
// grid (64 bn, 16 i-blocks of 64), block 128; each wave handles 16 query rows.
// ---------------------------------------------------------------------------
__global__ __launch_bounds__(128) void attn_kernel(
    const bf16* __restrict__ qaB, const bf16* __restrict__ kb,
    const bf16* __restrict__ vT, const bf16* __restrict__ Cbd,
    const unsigned char* __restrict__ mask, bf16* __restrict__ attn_vec) {
  __shared__ __align__(16) bf16 Pbuf[4][16][48];

  const int bn = blockIdx.x;
  const int b = bn >> 4, n = bn & 15;
  const int w = threadIdx.x >> 5;
  const int lane = threadIdx.x & 31;
  const int i0 = blockIdx.y * 64 + w * 16;
  const int lan15 = lane & 15;
  const int offA = (lane < 16) ? 0 : 16;
  const int offB = (lane < 16) ? 0 : 16;
  const int roff = (lane < 16) ? 0 : 8;

  const bf16* qrow = qaB + ((size_t)(i0 + lan15) * B_SZ + b) * D_MODEL + n * D_HEAD;
  v16bf aq0 = load_a_frag(qrow, offA);
  v16bf aq1 = load_a_frag(qrow + 32, offA);

  const bf16* cbase = Cbd + (size_t)bn * S_LEN * C_PITCH;
  const bf16* vbase = vT + (size_t)((b * N_HEAD + n) * D_HEAD) * (size_t)S_LEN;

  v8f O[4];
  float mrow[8], lrow[8];
#pragma unroll
  for (int t = 0; t < 4; ++t)
#pragma unroll
    for (int r = 0; r < 8; ++r) O[t][r] = 0.0f;
#pragma unroll
  for (int r = 0; r < 8; ++r) { mrow[r] = -3.0e38f; lrow[r] = 0.0f; }

  for (int j0 = 0; j0 < S_LEN; j0 += 32) {
    v8f s[2];
#pragma unroll
    for (int half = 0; half < 2; ++half) {
      int jb = j0 + half * 16;
      const bf16* krow =
          kb + ((size_t)(jb + lan15) * B_SZ + b) * D_MODEL + n * D_HEAD;
      v8f acc;
#pragma unroll
      for (int r = 0; r < 8; ++r) acc[r] = 0.0f;
      acc = wmma_bf(aq0, load_b_frag(krow + offB), acc);
      acc = wmma_bf(aq1, load_b_frag(krow + 32 + offB), acc);
      int j = jb + lan15;
#pragma unroll
      for (int r = 0; r < 8; ++r) {
        int i = i0 + r + roff;
        float bdv;
        if (j <= i)
          bdv = (float)cbase[(size_t)i * C_PITCH + (S_LEN + j - i)];
        else
          bdv = (float)cbase[(size_t)(i + 1) * C_PITCH + (j - i - 1)];
        float mv = mask[(size_t)i * S_LEN + j] ? 1.0f : 0.0f;
        acc[r] = (acc[r] + bdv) * 0.125f - 1.0e30f * mv;
      }
      s[half] = acc;
    }
    // online softmax update (rows live in 16-lane halves)
#pragma unroll
    for (int r = 0; r < 8; ++r) {
      float tmax = fmaxf(s[0][r], s[1][r]);
      tmax = fmaxf(tmax, __shfl_xor(tmax, 1, 32));
      tmax = fmaxf(tmax, __shfl_xor(tmax, 2, 32));
      tmax = fmaxf(tmax, __shfl_xor(tmax, 4, 32));
      tmax = fmaxf(tmax, __shfl_xor(tmax, 8, 32));
      float mn = fmaxf(mrow[r], tmax);
      float alpha = __expf(mrow[r] - mn);
      float p0 = __expf(s[0][r] - mn);
      float p1 = __expf(s[1][r] - mn);
      s[0][r] = p0;
      s[1][r] = p1;
      float ps = p0 + p1;
      ps += __shfl_xor(ps, 1, 32);
      ps += __shfl_xor(ps, 2, 32);
      ps += __shfl_xor(ps, 4, 32);
      ps += __shfl_xor(ps, 8, 32);
      lrow[r] = lrow[r] * alpha + ps;
      mrow[r] = mn;
      O[0][r] *= alpha;
      O[1][r] *= alpha;
      O[2][r] *= alpha;
      O[3][r] *= alpha;
    }
    // transpose P (C-layout -> A-layout) through LDS
#pragma unroll
    for (int r = 0; r < 8; ++r) {
      Pbuf[w][r + roff][lan15] = (bf16)s[0][r];
      Pbuf[w][r + roff][16 + lan15] = (bf16)s[1][r];
    }
    asm volatile("s_wait_dscnt 0" ::: "memory");
    v16bf aP = load_a_frag(&Pbuf[w][lan15][0], offA);
#pragma unroll
    for (int nt = 0; nt < 4; ++nt) {
      const bf16* vrow = vbase + (size_t)(nt * 16 + lan15) * S_LEN + j0;
      O[nt] = wmma_bf(aP, load_b_frag(vrow + offB), O[nt]);
    }
    asm volatile("s_wait_dscnt 0" ::: "memory");  // reads done before next store
  }
  // normalize + store
#pragma unroll
  for (int nt = 0; nt < 4; ++nt) {
#pragma unroll
    for (int r = 0; r < 8; ++r) {
      int i = i0 + r + roff;
      float v = O[nt][r] / lrow[r];
      attn_vec[((size_t)i * B_SZ + b) * D_MODEL + n * D_HEAD + nt * 16 + lan15] =
          (bf16)v;
    }
  }
}

// ---------------------------------------------------------------------------
// add + LayerNorm over D=1024; optional fp32 and bf16 outputs.
// ---------------------------------------------------------------------------
__global__ __launch_bounds__(256) void add_ln_kernel(
    const float* __restrict__ x1, const float* __restrict__ x2,
    const float* __restrict__ g, const float* __restrict__ beta,
    float* __restrict__ outf, bf16* __restrict__ outb) {
  __shared__ float sh0[256];
  __shared__ float sh1[256];
  const int row = blockIdx.x;
  const int tid = threadIdx.x;
  const float* p1 = x1 + (size_t)row * D_MODEL;
  const float* p2 = x2 + (size_t)row * D_MODEL;
  float v[4], s = 0.0f, s2 = 0.0f;
#pragma unroll
  for (int t = 0; t < 4; ++t) {
    float a = p1[tid + t * 256] + p2[tid + t * 256];
    v[t] = a;
    s += a;
    s2 += a * a;
  }
  sh0[tid] = s;
  sh1[tid] = s2;
  __syncthreads();
  for (int st = 128; st > 0; st >>= 1) {
    if (tid < st) {
      sh0[tid] += sh0[tid + st];
      sh1[tid] += sh1[tid + st];
    }
    __syncthreads();
  }
  float mean = sh0[0] * (1.0f / D_MODEL);
  float var = sh1[0] * (1.0f / D_MODEL) - mean * mean;
  float rstd = rsqrtf(var + 1.0e-5f);
#pragma unroll
  for (int t = 0; t < 4; ++t) {
    int d = tid + t * 256;
    float o = (v[t] - mean) * rstd * g[d] + beta[d];
    if (outf) outf[(size_t)row * D_MODEL + d] = o;
    if (outb) outb[(size_t)row * D_MODEL + d] = (bf16)o;
  }
}

// ---------------------------------------------------------------------------
// Host side
// ---------------------------------------------------------------------------
extern "C" void kernel_launch(void* const* d_in, const int* in_sizes, int n_in,
                              void* d_out, int out_size, void* d_ws,
                              size_t ws_size, hipStream_t stream) {
  (void)in_sizes; (void)n_in; (void)out_size; (void)ws_size;
  const float* src = (const float*)d_in[0];
  const float* pos = (const float*)d_in[1];
  const unsigned char* mask = (const unsigned char*)d_in[2];
  const float* qw = (const float*)d_in[3];
  const float* kw = (const float*)d_in[4];
  const float* vw = (const float*)d_in[5];
  const float* rw = (const float*)d_in[6];
  const float* ow = (const float*)d_in[7];
  const float* r_w_bias = (const float*)d_in[8];
  const float* r_r_bias = (const float*)d_in[9];
  const float* w1 = (const float*)d_in[10];
  const float* b1 = (const float*)d_in[11];
  const float* w2 = (const float*)d_in[12];
  const float* b2 = (const float*)d_in[13];
  const float* g1 = (const float*)d_in[14];
  const float* be1 = (const float*)d_in[15];
  const float* g2 = (const float*)d_in[16];
  const float* be2 = (const float*)d_in[17];

  char* cur = (char*)d_ws;
  auto carve = [&](size_t bytes) -> void* {
    void* p = (void*)cur;
    cur += (bytes + 255) & ~(size_t)255;
    return p;
  };
  bf16* srcb = (bf16*)carve((size_t)M_ROWS * D_MODEL * 2);
  bf16* posb = (bf16*)carve((size_t)MP_ROWS * D_MODEL * 2);
  bf16* qwb = (bf16*)carve((size_t)D_MODEL * 1024 * 2);
  bf16* kwb = (bf16*)carve((size_t)D_MODEL * 1024 * 2);
  bf16* vwb = (bf16*)carve((size_t)D_MODEL * 1024 * 2);
  bf16* rwbw = (bf16*)carve((size_t)D_MODEL * 1024 * 2);
  bf16* owT = (bf16*)carve((size_t)D_MODEL * 1024 * 2);
  bf16* w1T = (bf16*)carve((size_t)D_MODEL * D_FF * 2);
  bf16* w2T = (bf16*)carve((size_t)D_FF * D_MODEL * 2);
  bf16* qaB = (bf16*)carve((size_t)M_ROWS * 1024 * 2);
  bf16* qbB = (bf16*)carve((size_t)M_ROWS * 1024 * 2);
  bf16* kbuf = (bf16*)carve((size_t)M_ROWS * 1024 * 2);
  bf16* vT = (bf16*)carve((size_t)M_ROWS * 1024 * 2);
  bf16* kposb = (bf16*)carve((size_t)MP_ROWS * 1024 * 2);
  bf16* Cbd = (bf16*)carve((size_t)64 * S_LEN * C_PITCH * 2);
  bf16* attnv = (bf16*)carve((size_t)M_ROWS * 1024 * 2);
  float* attn_out = (float*)carve((size_t)M_ROWS * D_MODEL * 4);
  float* xf = (float*)carve((size_t)M_ROWS * D_MODEL * 4);
  bf16* xb = (bf16*)carve((size_t)M_ROWS * D_MODEL * 2);
  bf16* hb = (bf16*)carve((size_t)M_ROWS * D_FF * 2);
  float* ff = (float*)carve((size_t)M_ROWS * D_MODEL * 4);

  // --- conversions ---
  cvt_kernel<<<(M_ROWS * D_MODEL + 1023) / 1024, 256, 0, stream>>>(
      src, srcb, M_ROWS * D_MODEL);
  cvt_kernel<<<(MP_ROWS * D_MODEL + 1023) / 1024, 256, 0, stream>>>(
      pos, posb, MP_ROWS * D_MODEL);
  cvt_kernel<<<1024, 256, 0, stream>>>(qw, qwb, 1048576);
  cvt_kernel<<<1024, 256, 0, stream>>>(kw, kwb, 1048576);
  cvt_kernel<<<1024, 256, 0, stream>>>(vw, vwb, 1048576);
  cvt_kernel<<<1024, 256, 0, stream>>>(rw, rwbw, 1048576);
  cvtT_kernel<<<4096, 256, 0, stream>>>(ow, owT, 1024, 1024);     // [k=(n,d)][h]
  cvtT_kernel<<<16384, 256, 0, stream>>>(w1, w1T, D_FF, D_MODEL); // [h][f]
  cvtT_kernel<<<16384, 256, 0, stream>>>(w2, w2T, D_MODEL, D_FF); // [f][d]

  // --- projections ---
  dim3 blk(256);
  gemm_bf16_kernel<3><<<dim3(8, 32), blk, 0, stream>>>(
      srcb, qwb, M_ROWS, 1024, 1024, r_w_bias, r_r_bias, qaB, qbB);
  gemm_bf16_kernel<1><<<dim3(8, 32), blk, 0, stream>>>(
      srcb, kwb, M_ROWS, 1024, 1024, nullptr, nullptr, kbuf, nullptr);
  gemm_bf16_kernel<4><<<dim3(8, 32), blk, 0, stream>>>(
      srcb, vwb, M_ROWS, 1024, 1024, nullptr, nullptr, vT, nullptr);
  gemm_bf16_kernel<1><<<dim3(8, 33), blk, 0, stream>>>(
      posb, rwbw, MP_ROWS, 1024, 1024, nullptr, nullptr, kposb, nullptr);

  // --- raw bd matrix + flash attention ---
  bd_kernel<<<dim3(64, 16, 17), dim3(128), 0, stream>>>(qbB, kposb, Cbd);
  attn_kernel<<<dim3(64, 16), dim3(128), 0, stream>>>(qaB, kbuf, vT, Cbd, mask,
                                                      attnv);

  // --- output projection + LN1 ---
  gemm_bf16_kernel<0><<<dim3(8, 32), blk, 0, stream>>>(
      attnv, owT, M_ROWS, 1024, 1024, nullptr, nullptr, attn_out, nullptr);
  add_ln_kernel<<<M_ROWS, 256, 0, stream>>>(src, attn_out, g1, be1, xf, xb);

  // --- FFN + LN2 ---
  gemm_bf16_kernel<2><<<dim3(32, 32), blk, 0, stream>>>(
      xb, w1T, M_ROWS, D_FF, 1024, b1, nullptr, hb, nullptr);
  gemm_bf16_kernel<0><<<dim3(8, 32), blk, 0, stream>>>(
      hb, w2T, M_ROWS, 1024, D_FF, b2, nullptr, ff, nullptr);
  add_ln_kernel<<<M_ROWS, 256, 0, stream>>>(xf, ff, g2, be2, (float*)d_out,
                                            nullptr);
}
